// PointnetSAModuleVotes_38044820308009
// MI455X (gfx1250) — matrix operand order repaired and checked by reading, not compiled, not run on hardware
//
#include <hip/hip_runtime.h>
#include <hip/hip_bf16.h>
#include <math.h>

typedef __bf16 bf16_t;
typedef __attribute__((ext_vector_type(16))) __bf16 v16bf;
typedef __attribute__((ext_vector_type(8)))  __bf16 v8bf;
typedef __attribute__((ext_vector_type(8)))  float  v8f;

#define B_   8
#define N_   16384
#define C_   16
#define NP_  1024
#define NS_  64
#define RAD2 0.04f
#define MCOLS (B_*NP_*NS_)          /* 524288 GEMM columns */

/* d_out offsets (float elements), outputs concatenated in return order */
#define OUT_XYZ  0                  /* (8,1024,3)   = 24576  */
#define OUT_FEAT 24576              /* (8,128,1024) = 1048576*/
#define OUT_INDS 1073152            /* (8,1024)     = 8192   */
#define OUT_POSE 1081344            /* (8,1024)     = 8192   */

/* workspace byte offsets */
#define WS_STATS 0                         /* 512 f32: 3 layers of (sum,sumsq)        */
#define WS_SS    2048                      /* 512 f32: 3 layers of (scale,shift)      */
#define WS_WP1   4096                      /* packed W1 fragments: 4t*1kk*512 bf16    */
#define WS_WP2   8192                      /* packed W2 fragments: 4t*2kk*512 bf16    */
#define WS_WP3   16384                     /* packed W3 fragments: 8t*2kk*512 bf16    */
#define WS_X0    32768                     /* bf16 [524288][32]  = 32 MB  */
#define WS_Y1    (WS_X0 + 33554432)        /* bf16 [524288][64]  = 64 MB  */
#define WS_Y2    (WS_Y1 + 67108864)        /* bf16 [524288][64]  = 64 MB  */
#define WS_Y3    (WS_Y2 + 67108864)        /* bf16 [524288][128] = 128 MB */

/* ---------------- FPS: one block per batch, dist state in registers ---------------- */
__global__ __launch_bounds__(1024) void fps_kernel(const float* __restrict__ xyz,
                                                   float* __restrict__ out) {
  int b = blockIdx.x;
  int t = threadIdx.x;
  const float* xb = xyz + (size_t)b * N_ * 3;
  float dloc[16];
#pragma unroll
  for (int j = 0; j < 16; ++j) dloc[j] = 1e10f;

  __shared__ float sv[32];
  __shared__ int   si[32];
  __shared__ float cent[3];
  __shared__ int   scur;

  int lane = t & 31, wid = t >> 5;
  int cur = 0;
  for (int it = 0; it < NP_; ++it) {
    if (t == 0) {
      float cx = xb[cur * 3 + 0], cy = xb[cur * 3 + 1], cz = xb[cur * 3 + 2];
      cent[0] = cx; cent[1] = cy; cent[2] = cz;
      out[OUT_XYZ + (b * NP_ + it) * 3 + 0] = cx;
      out[OUT_XYZ + (b * NP_ + it) * 3 + 1] = cy;
      out[OUT_XYZ + (b * NP_ + it) * 3 + 2] = cz;
      out[OUT_INDS + b * NP_ + it] = (float)cur;
    }
    __syncthreads();
    float cx = cent[0], cy = cent[1], cz = cent[2];
    float bv = -1.0f; int bi = 0;
#pragma unroll
    for (int j = 0; j < 16; ++j) {
      int i = j * 1024 + t;
      float dx = xb[i * 3 + 0] - cx;
      float dy = xb[i * 3 + 1] - cy;
      float dz = xb[i * 3 + 2] - cz;
      float d = dx * dx + dy * dy + dz * dz;
      float nd = fminf(dloc[j], d);
      dloc[j] = nd;
      if (nd > bv) { bv = nd; bi = i; }
    }
    for (int off = 16; off > 0; off >>= 1) {
      float ov = __shfl_xor(bv, off, 32);
      int   oi = __shfl_xor(bi, off, 32);
      if (ov > bv || (ov == bv && oi < bi)) { bv = ov; bi = oi; }
    }
    if (lane == 0) { sv[wid] = bv; si[wid] = bi; }
    __syncthreads();
    if (wid == 0) {
      float v = sv[lane]; int i2 = si[lane];
      for (int off = 16; off > 0; off >>= 1) {
        float ov = __shfl_xor(v, off, 32);
        int   oi = __shfl_xor(i2, off, 32);
        if (ov > v || (ov == v && oi < i2)) { v = ov; i2 = oi; }
      }
      if (lane == 0) scur = i2;
    }
    __syncthreads();
    cur = scur;
  }
}

/* ------------- ball query + pose + canonical rotation + bf16 GEMM input ------------- */
__global__ __launch_bounds__(256) void group_kernel(const float* __restrict__ xyz,
                                                    const float* __restrict__ feats,
                                                    const float* __restrict__ pose,
                                                    const float* __restrict__ pmask,
                                                    float* __restrict__ out,
                                                    bf16_t* __restrict__ X0) {
  __shared__ int idxbuf[8][64];
  int tid = threadIdx.x, lane = tid & 31, w = tid >> 5;
  int q = blockIdx.x * 8 + w;          /* 0..8191 */
  int b = q >> 10;
  const float* xb = xyz + (size_t)b * N_ * 3;
  float cx = out[OUT_XYZ + q * 3 + 0];
  float cy = out[OUT_XYZ + q * 3 + 1];
  float cz = out[OUT_XYZ + q * 3 + 2];

  int cnt = 0;
  for (int base = 0; base < N_; base += 32) {
    int i = base + lane;
    float dx = xb[i * 3 + 0] - cx;
    float dy = xb[i * 3 + 1] - cy;
    float dz = xb[i * 3 + 2] - cz;
    bool hit = (dx * dx + dy * dy + dz * dz) < RAD2;
    unsigned m = (unsigned)__ballot(hit);
    int pos = cnt + __popc(m & ((1u << lane) - 1u));
    if (hit && pos < NS_) idxbuf[w][pos] = i;
    cnt += __popc(m);
    if (cnt >= NS_) break;            /* uniform across wave */
  }
  __syncthreads();
  int firstIdx = (cnt > 0) ? idxbuf[w][0] : 0;
  int total = cnt < NS_ ? cnt : NS_;
  if (cnt == 0) total = 0;
  for (int s = total + lane; s < NS_; s += 32) idxbuf[w][s] = firstIdx;
  __syncthreads();

  /* circular-mean pose over FG points (includes pad duplicates, like reference) */
  float ssum = 0.f, csum = 0.f, msum = 0.f;
  for (int s = lane; s < NS_; s += 32) {
    int i = idxbuf[w][s];
    float mk = pmask[b * N_ + i], ph = pose[b * N_ + i];
    ssum += mk * __sinf(ph);
    csum += mk * __cosf(ph);
    msum += mk;
  }
  for (int off = 16; off > 0; off >>= 1) {
    ssum += __shfl_xor(ssum, off, 32);
    csum += __shfl_xor(csum, off, 32);
    msum += __shfl_xor(msum, off, 32);
  }
  float mp = (msum > 0.f) ? atan2f(ssum, csum) : 0.f;
  if (lane == 0) out[OUT_POSE + q] = mp;
  float ct = __cosf(mp), st = __sinf(mp);

  for (int s = lane; s < NS_; s += 32) {
    int i = idxbuf[w][s];
    float rx = xb[i * 3 + 0] - cx;
    float ry = xb[i * 3 + 1] - cy;
    float rz = xb[i * 3 + 2] - cz;
    float xr =  ct * rx + st * ry;
    float yr = -st * rx + ct * ry;
    bf16_t* row = X0 + (size_t)(q * NS_ + s) * 32;
    row[0] = (bf16_t)xr; row[1] = (bf16_t)yr; row[2] = (bf16_t)rz;
#pragma unroll
    for (int c = 0; c < 16; ++c)
      row[3 + c] = (bf16_t)feats[((size_t)b * C_ + c) * N_ + i];
#pragma unroll
    for (int c = 19; c < 32; ++c) row[c] = (bf16_t)0.f;
  }
}

/* ------------- pack weights into per-lane WMMA A-fragment layout (bf16) ------------- */
__global__ void packw_kernel(const float* __restrict__ W, int WCin, int NT, int KK,
                             bf16_t* __restrict__ dst) {
  int totalE = NT * KK * 512;
  for (int e = blockIdx.x * blockDim.x + threadIdx.x; e < totalE;
       e += gridDim.x * blockDim.x) {
    int h = e & 15;
    int lane = (e >> 4) & 31;
    int rest = e >> 9;
    int t2 = rest % NT;
    int kk2 = rest / NT;
    int row = t2 * 16 + (lane & 15);
    int hi = lane >> 4;
    int c = kk2 * 32 + hi * 16 + h;   /* contiguous k order: k = 16*hi + h */
    float v = (c < WCin) ? W[row * WCin + c] : 0.f;
    dst[e] = (bf16_t)v;
  }
}

/* --------------- GEMM + (optional fused BN+ReLU on input) + stats --------------- */
template <int CIN, int COUT, bool BN>
__global__ __launch_bounds__(256) void gemm_kernel(const bf16_t* __restrict__ Xin,
                                                   const bf16_t* __restrict__ Wp,
                                                   const float* __restrict__ bias,
                                                   const float* __restrict__ ss,
                                                   bf16_t* __restrict__ Yout,
                                                   float* __restrict__ stats) {
  constexpr int NT = COUT / 16;
  constexpr int KK = CIN / 32;
  __shared__ float sStat[2 * COUT];
  int tid = threadIdx.x;
  for (int i = tid; i < 2 * COUT; i += 256) sStat[i] = 0.f;
  __syncthreads();

  int lane = tid & 31, w = tid >> 5;
  int col = lane & 15, hi = lane >> 4;
  int m = blockIdx.x * 128 + w * 16 + col;

  v8f acc[NT];
#pragma unroll
  for (int t = 0; t < NT; ++t)
#pragma unroll
    for (int r = 0; r < 8; ++r) acc[t][r] = 0.f;

#pragma unroll
  for (int kk2 = 0; kk2 < KK; ++kk2) {
    int kbase = kk2 * 32 + hi * 16;
    v16bf bfrag;
    const v16bf raw = *(const v16bf*)(Xin + (size_t)m * CIN + kbase);
    if constexpr (BN) {
#pragma unroll
      for (int e = 0; e < 16; ++e) {
        int c = kbase + e;
        float f = ss[c] * (float)raw[e] + ss[CIN + c];
        f = fmaxf(f, 0.f);
        bfrag[e] = (bf16_t)f;
      }
    } else {
      bfrag = raw;
    }
#pragma unroll
    for (int t = 0; t < NT; ++t) {
      v16bf afrag = *(const v16bf*)(Wp + ((size_t)(kk2 * NT + t) * 32 + lane) * 16);
      acc[t] = __builtin_amdgcn_wmma_f32_16x16x32_bf16(
          false, afrag, false, bfrag, (short)0, acc[t], false, false);
    }
  }

#pragma unroll
  for (int t = 0; t < NT; ++t) {
    int cb = t * 16 + hi * 8;
    v8bf pk;
#pragma unroll
    for (int r = 0; r < 8; ++r) {
      float v = acc[t][r] + bias[cb + r];
      acc[t][r] = v;
      pk[r] = (bf16_t)v;
    }
    *(v8bf*)(Yout + (size_t)m * COUT + cb) = pk;
    /* per-channel sum / sumsq: reduce across the 16 columns of this half-wave */
#pragma unroll
    for (int r = 0; r < 8; ++r) {
      float s1 = acc[t][r], s2 = s1 * s1;
      for (int off = 8; off > 0; off >>= 1) {
        s1 += __shfl_xor(s1, off, 32);
        s2 += __shfl_xor(s2, off, 32);
      }
      if (col == 0) {
        atomicAdd(&sStat[cb + r], s1);
        atomicAdd(&sStat[COUT + cb + r], s2);
      }
    }
  }
  __syncthreads();
  for (int i = tid; i < 2 * COUT; i += 256) atomicAdd(&stats[i], sStat[i]);
}

__global__ void zero_kernel(float* __restrict__ stats) {
  stats[threadIdx.x] = 0.f;     /* <<<1,512>>> */
}

__global__ void finalize_kernel(const float* __restrict__ stats,
                                const float* __restrict__ gamma,
                                const float* __restrict__ beta,
                                float* __restrict__ ss, int Cout) {
  int c = threadIdx.x;
  if (c >= Cout) return;
  float inv = 1.0f / (float)MCOLS;
  float mu = stats[c] * inv;
  float var = stats[Cout + c] * inv - mu * mu;
  float sc = gamma[c] * rsqrtf(var + 1e-5f);
  ss[c] = sc;
  ss[Cout + c] = beta[c] - mu * sc;
}

/* --------------- final BN + ReLU + max over samples -> new_features --------------- */
__global__ __launch_bounds__(256) void maxpool_kernel(const bf16_t* __restrict__ Y3,
                                                      const float* __restrict__ ss,
                                                      float* __restrict__ out) {
  int g = blockIdx.x * 256 + threadIdx.x;  /* 8*1024*128 = 1048576 */
  int o = g & 127;
  int pp = g >> 7;                         /* b*1024 + p */
  float sc = ss[o], sh = ss[128 + o];
  float mx = 0.f;                          /* relu floor */
  const bf16_t* base = Y3 + (size_t)pp * 64 * 128 + o;
  for (int s = 0; s < 64; ++s) {
    float v = sc * (float)base[s * 128] + sh;
    mx = fmaxf(mx, v);
  }
  int b = pp >> 10, p = pp & 1023;
  out[OUT_FEAT + ((size_t)b * 128 + o) * 1024 + p] = mx;
}

extern "C" void kernel_launch(void* const* d_in, const int* in_sizes, int n_in,
                              void* d_out, int out_size, void* d_ws, size_t ws_size,
                              hipStream_t stream) {
  const float* xyz   = (const float*)d_in[0];
  const float* feats = (const float*)d_in[1];
  const float* pose  = (const float*)d_in[2];
  const float* pmask = (const float*)d_in[3];
  const float* W1 = (const float*)d_in[4];  const float* b1 = (const float*)d_in[5];
  const float* g1 = (const float*)d_in[6];  const float* e1 = (const float*)d_in[7];
  const float* W2 = (const float*)d_in[8];  const float* b2 = (const float*)d_in[9];
  const float* g2 = (const float*)d_in[10]; const float* e2 = (const float*)d_in[11];
  const float* W3 = (const float*)d_in[12]; const float* b3 = (const float*)d_in[13];
  const float* g3 = (const float*)d_in[14]; const float* e3 = (const float*)d_in[15];

  char* ws = (char*)d_ws;
  float*  stats = (float*)(ws + WS_STATS);
  float*  ssb   = (float*)(ws + WS_SS);
  bf16_t* wp1   = (bf16_t*)(ws + WS_WP1);
  bf16_t* wp2   = (bf16_t*)(ws + WS_WP2);
  bf16_t* wp3   = (bf16_t*)(ws + WS_WP3);
  bf16_t* X0    = (bf16_t*)(ws + WS_X0);
  bf16_t* Y1    = (bf16_t*)(ws + WS_Y1);
  bf16_t* Y2    = (bf16_t*)(ws + WS_Y2);
  bf16_t* Y3    = (bf16_t*)(ws + WS_Y3);
  float* out = (float*)d_out;

  zero_kernel<<<1, 512, 0, stream>>>(stats);
  packw_kernel<<<8,  256, 0, stream>>>(W1, 19, 4, 1, wp1);
  packw_kernel<<<16, 256, 0, stream>>>(W2, 64, 4, 2, wp2);
  packw_kernel<<<32, 256, 0, stream>>>(W3, 64, 8, 2, wp3);

  fps_kernel<<<8, 1024, 0, stream>>>(xyz, out);
  group_kernel<<<1024, 256, 0, stream>>>(xyz, feats, pose, pmask, out, X0);

  gemm_kernel<32, 64, false><<<4096, 256, 0, stream>>>(X0, wp1, b1, nullptr, Y1, stats + 0);
  finalize_kernel<<<1, 128, 0, stream>>>(stats + 0, g1, e1, ssb + 0, 64);

  gemm_kernel<64, 64, true><<<4096, 256, 0, stream>>>(Y1, wp2, b2, ssb + 0, Y2, stats + 128);
  finalize_kernel<<<1, 128, 0, stream>>>(stats + 128, g2, e2, ssb + 128, 64);

  gemm_kernel<64, 128, true><<<4096, 256, 0, stream>>>(Y2, wp3, b3, ssb + 128, Y3, stats + 256);
  finalize_kernel<<<1, 128, 0, stream>>>(stats + 256, g3, e3, ssb + 256, 128);

  maxpool_kernel<<<4096, 256, 0, stream>>>(Y3, ssb + 256, out);
}
